// Community_652835029417
// MI455X (gfx1250) — matrix operand order repaired and checked by reading, not compiled, not run on hardware
//
#include <hip/hip_runtime.h>
#include <math.h>

#define T_STEPS 64
#define BATCH   512
#define NA      4
#define N_IN    128
#define N_HID   512
#define N_OUT   10
#define OUT_SCALE 0.1f

typedef __attribute__((ext_vector_type(16))) __bf16 v16bf;
typedef __attribute__((ext_vector_type(8)))  float  v8f;

struct alignas(16) B16 { unsigned int x[4]; };   // one 16-byte global_load_b128

// ---- WMMA fragment loaders (bf16, row-major, K-contiguous) ------------------
// A (16x32): lane l -> row m0+(l&15); elems 0..7 = K kb+8*(l>>4)+0..7, elems 8..15 = +16
__device__ __forceinline__ v16bf load_a_frag(const __bf16* base, int ld, int row,
                                             int kbase, int lh) {
  v16bf f;
  const __bf16* p = base + (size_t)row * ld + kbase + lh * 8;
  reinterpret_cast<B16*>(&f)[0] = *reinterpret_cast<const B16*>(p);
  reinterpret_cast<B16*>(&f)[1] = *reinterpret_cast<const B16*>(p + 16);
  return f;
}
// B (32x16): lane l -> col n0+(l&15); elems 0..7 = K kb+8*(l>>4)+0..7, elems 8..15 = +16
__device__ __forceinline__ v16bf load_b_frag(const __bf16* base, int ld, int nrow,
                                             int kbase, int lh) {
  v16bf f;
  const __bf16* p = base + (size_t)nrow * ld + kbase + lh * 8;
  reinterpret_cast<B16*>(&f)[0] = *reinterpret_cast<const B16*>(p);
  reinterpret_cast<B16*>(&f)[1] = *reinterpret_cast<const B16*>(p + 16);
  return f;
}

// One K-segment of the per-agent GEMM: acc(32x32) += A(32xK) * B(KxN=32)^T-layout
__device__ __forceinline__ void gemm_seg(v8f (&acc)[2][2],
                                         const __bf16* __restrict__ Ab, int lda,
                                         const __bf16* __restrict__ Bb, int ldb,
                                         int K, int m0, int n0, int lq, int lh) {
  for (int k = 0; k < K; k += 32) {
    v16bf a0 = load_a_frag(Ab, lda, m0 + lq,      k, lh);
    v16bf a1 = load_a_frag(Ab, lda, m0 + 16 + lq, k, lh);
    v16bf b0 = load_b_frag(Bb, ldb, n0 + lq,      k, lh);
    v16bf b1 = load_b_frag(Bb, ldb, n0 + 16 + lq, k, lh);
    acc[0][0] = __builtin_amdgcn_wmma_f32_16x16x32_bf16(false, a0, false, b0, (short)0, acc[0][0], false, false);
    acc[0][1] = __builtin_amdgcn_wmma_f32_16x16x32_bf16(false, a0, false, b1, (short)0, acc[0][1], false, false);
    acc[1][0] = __builtin_amdgcn_wmma_f32_16x16x32_bf16(false, a1, false, b0, (short)0, acc[1][0], false, false);
    acc[1][1] = __builtin_amdgcn_wmma_f32_16x16x32_bf16(false, a1, false, b1, (short)0, acc[1][1], false, false);
  }
}

// ---- one recurrence step: pre = comm + x@Win^T + h@Whh^T + biases; h=tanh ---
// grid (8 n-tiles, 8 m-tiles, 4 agents), 128 threads = 4 waves, wave tile 32x32
__global__ __launch_bounds__(128) void k_step(
    const __bf16* __restrict__ xb,    // (T,B,N_IN) bf16
    const __bf16* __restrict__ win,   // (A,N_HID,N_IN)  N-major, K-contig
    const __bf16* __restrict__ whh,   // (A,N_HID,N_HID) N-major, K-contig
    const __bf16* __restrict__ wm,    // (j,i,o,h) bf16, mask & OUT_SCALE folded
    const float*  __restrict__ bin, const float* __restrict__ bhh,
    const __bf16* __restrict__ h_cur, // (A,B,N_HID)
    __bf16* __restrict__ h_nxt, float* __restrict__ h_f32,
    float* __restrict__ conn, int t)
{
  const int lane = threadIdx.x & 31;
  const int wv   = threadIdx.x >> 5;
  const int lq = lane & 15, lh = lane >> 4;
  const int a  = blockIdx.z;
  const int m0 = blockIdx.y * 64 + (wv & 1) * 32;   // batch rows
  const int n0 = blockIdx.x * 64 + (wv >> 1) * 32;  // hidden cols

  v8f acc[2][2] = {};

  // Phase 1: comm[a] = sum_j h[j] @ (Wm[j,a] * mask * OUT_SCALE)   (zero at t=0 since h=0)
  for (int j = 0; j < NA; ++j)
    gemm_seg(acc, h_cur + (size_t)j * BATCH * N_HID, N_HID,
             wm + ((size_t)j * NA + a) * N_HID * N_HID, N_HID,
             N_HID, m0, n0, lq, lh);

  if (t > 0) {  // connections output records comm for t >= 1
    float* cb = conn + ((size_t)(t - 1) * NA + a) * (size_t)BATCH * N_HID;
    #pragma unroll
    for (int im = 0; im < 2; ++im)
      #pragma unroll
      for (int in = 0; in < 2; ++in) {
        const int nn = n0 + 16 * in + lq;
        const int mb = m0 + 16 * im + lh * 8;
        #pragma unroll
        for (int r = 0; r < 8; ++r)
          cb[(size_t)(mb + r) * N_HID + nn] = acc[im][in][r];
      }
  }

  // Phase 2: += x_t @ W_in[a]^T   (K = 128)
  gemm_seg(acc, xb + (size_t)t * BATCH * N_IN, N_IN,
           win + (size_t)a * N_HID * N_IN, N_IN, N_IN, m0, n0, lq, lh);
  // Phase 3: += h[a] @ W_hh[a]^T  (K = 512)
  gemm_seg(acc, h_cur + (size_t)a * BATCH * N_HID, N_HID,
           whh + (size_t)a * N_HID * N_HID, N_HID, N_HID, m0, n0, lq, lh);

  // bias + tanh + store (f32 for readout/h_final, bf16 for next step's WMMA)
  #pragma unroll
  for (int in = 0; in < 2; ++in) {
    const int nn = n0 + 16 * in + lq;
    const float bias = bin[a * N_HID + nn] + bhh[a * N_HID + nn];
    #pragma unroll
    for (int im = 0; im < 2; ++im) {
      const int mb = m0 + 16 * im + lh * 8;
      #pragma unroll
      for (int r = 0; r < 8; ++r) {
        float v = tanhf(acc[im][in][r] + bias);
        size_t idx = ((size_t)a * BATCH + mb + r) * N_HID + nn;
        h_f32[idx] = v;
        h_nxt[idx] = (__bf16)v;
      }
    }
  }
}

// ---- tiny readout: out[b,o] = sum_c h[c>>9, b, c&511] * Wro[o,c] + bro[o] ---
__global__ __launch_bounds__(256) void k_readout(const float* __restrict__ hf,
                                                 const float* __restrict__ wro,
                                                 const float* __restrict__ bro,
                                                 float* __restrict__ out) {
  __shared__ float sm[256];
  const int b = blockIdx.x, tid = threadIdx.x;
  float p[N_OUT] = {};
  for (int c = tid; c < NA * N_HID; c += 256) {
    float v = hf[(((size_t)(c >> 9)) * BATCH + b) * N_HID + (c & (N_HID - 1))];
    #pragma unroll
    for (int o = 0; o < N_OUT; ++o) p[o] += v * wro[o * (NA * N_HID) + c];
  }
  for (int o = 0; o < N_OUT; ++o) {
    sm[tid] = p[o]; __syncthreads();
    for (int s = 128; s > 0; s >>= 1) { if (tid < s) sm[tid] += sm[tid + s]; __syncthreads(); }
    if (tid == 0) out[b * N_OUT + o] = sm[0] + bro[o];
    __syncthreads();
  }
}

// ---- one-time prep: f32 -> bf16, and Wm transpose+mask+scale fold -----------
__global__ void k_cvt(const float* __restrict__ in, __bf16* __restrict__ out, int n) {
  int i = blockIdx.x * blockDim.x + threadIdx.x;
  if (i < n) out[i] = (__bf16)in[i];
}
__global__ void k_prep_wm(const float* __restrict__ w, const unsigned char* __restrict__ mask,
                          __bf16* __restrict__ out) {
  int i = blockIdx.x * blockDim.x + threadIdx.x;          // dst index: (p, o, h)
  if (i >= NA * NA * N_HID * N_HID) return;
  int hh = i & (N_HID - 1);
  int o  = (i >> 9) & (N_HID - 1);
  int p  = i >> 18;                                       // p = j*NA + i_dst
  size_t src = ((size_t)p * N_HID + hh) * N_HID + o;      // source: (p, h, o)
  out[i] = (__bf16)(w[src] * (mask[src] ? OUT_SCALE : 0.0f));
}

// -----------------------------------------------------------------------------
extern "C" void kernel_launch(void* const* d_in, const int* in_sizes, int n_in,
                              void* d_out, int out_size, void* d_ws, size_t ws_size,
                              hipStream_t stream) {
  const float* x     = (const float*)d_in[0];
  const float* W_in  = (const float*)d_in[1];
  const float* b_in  = (const float*)d_in[2];
  const float* W_hh  = (const float*)d_in[3];
  const float* b_hh  = (const float*)d_in[4];
  const float* W_cm  = (const float*)d_in[5];
  const float* W_ro  = (const float*)d_in[6];
  const float* b_ro  = (const float*)d_in[7];
  const unsigned char* mask = (const unsigned char*)d_in[8];  // jax bool = 1 byte

  float* out_outputs = (float*)d_out;                              // (T,B,10)
  float* out_hfinal  = out_outputs + (size_t)T_STEPS * BATCH * N_OUT;
  float* out_conn    = out_hfinal + (size_t)NA * BATCH * N_HID;    // (T-1,A,B,H)

  // workspace carve-up (all 256B aligned)
  char* ws = (char*)d_ws;
  auto carve = [&](size_t bytes) { char* p = ws; ws += (bytes + 255) & ~(size_t)255; return p; };
  __bf16* xb   = (__bf16*)carve((size_t)T_STEPS * BATCH * N_IN * 2);
  __bf16* winb = (__bf16*)carve((size_t)NA * N_HID * N_IN * 2);
  __bf16* whhb = (__bf16*)carve((size_t)NA * N_HID * N_HID * 2);
  __bf16* wmb  = (__bf16*)carve((size_t)NA * NA * N_HID * N_HID * 2);
  __bf16* hb0  = (__bf16*)carve((size_t)NA * BATCH * N_HID * 2);
  __bf16* hb1  = (__bf16*)carve((size_t)NA * BATCH * N_HID * 2);
  float*  hf   = (float*) carve((size_t)NA * BATCH * N_HID * 4);
  __bf16* hb[2] = {hb0, hb1};

  // one-time prep (re-run every call; deterministic)
  const int nx  = T_STEPS * BATCH * N_IN;
  const int nwi = NA * N_HID * N_IN;
  const int nwh = NA * N_HID * N_HID;
  const int nwm = NA * NA * N_HID * N_HID;
  k_cvt<<<(nx  + 255) / 256, 256, 0, stream>>>(x,    xb,   nx);
  k_cvt<<<(nwi + 255) / 256, 256, 0, stream>>>(W_in, winb, nwi);
  k_cvt<<<(nwh + 255) / 256, 256, 0, stream>>>(W_hh, whhb, nwh);
  k_prep_wm<<<(nwm + 255) / 256, 256, 0, stream>>>(W_cm, mask, wmb);
  hipMemsetAsync(hb[0], 0, (size_t)NA * BATCH * N_HID * 2, stream);  // h0 = 0

  dim3 grid(N_HID / 64, BATCH / 64, NA);  // (8, 8, 4) blocks of 4 waves
  for (int t = 0; t < T_STEPS; ++t) {
    int cur = t & 1, nxt = cur ^ 1;
    k_step<<<grid, 128, 0, stream>>>(xb, winb, whhb, wmb, b_in, b_hh,
                                     hb[cur], hb[nxt], hf, out_conn, t);
    k_readout<<<BATCH, 256, 0, stream>>>(hf, W_ro, b_ro,
                                         out_outputs + (size_t)t * BATCH * N_OUT);
  }
  hipMemcpyAsync(out_hfinal, hf, (size_t)NA * BATCH * N_HID * sizeof(float),
                 hipMemcpyDeviceToDevice, stream);
}
